// TGCN_8856222564438
// MI455X (gfx1250) — matrix operand order repaired and checked by reading, not compiled, not run on hardware
//
#include <hip/hip_runtime.h>

// ---------------------------------------------------------------------------
// TGAT-style single-query attention, fused for MI455X (gfx1250, wave32, WMMA)
// B=8192, N=32, D=128, M=384, H=2, DK=192
// Key refactor: scores = k @ (w_k^T qh)_head ; out = (attn^T k) @ w_v^T_head
// -> activations (403 MB) are streamed from HBM exactly once (memory-bound,
//    ~18us floor @ 23.3 TB/s); all GEMMs run as bf16 v_wmma_f32_16x16x32_bf16.
// ---------------------------------------------------------------------------

typedef __bf16 bf16_t;
typedef __attribute__((ext_vector_type(16))) __bf16 v16bf;
typedef __attribute__((ext_vector_type(8)))  float  v8f;

union Frag { v16bf v; unsigned u[8]; };

static constexpr int B_ = 8192, N_ = 32, D_ = 128, M_ = 384, DK_ = 192;

// bf16 weight workspace layout (element offsets)
static constexpr int WQ_OFF  = 0;        // w_q   [384][384] (row j, col i)
static constexpr int WKT_OFF = 147456;   // w_k^T [i=384][j=384]
static constexpr int WV_OFF  = 294912;   // w_v   [384][384]
static constexpr int FCW_OFF = 442368;   // fc_w  [384][384]
static constexpr int MG1_OFF = 589824;   // mg_w1 [128][512]
static constexpr int MG2_OFF = 655360;   // mg_w2 [128][128]
static constexpr int WS_ELEMS = 671744;

// LDS strides (bf16 elems unless noted), all even so 32-bit pair loads align
static constexpr int KSTR  = 392;  // k chunk row stride
static constexpr int KTSTR = 34;   // k^T row stride
static constexpr int QSTR  = 392;  // qin / qh
static constexpr int USTR  = 384;  // u[b][h][i]
static constexpr int CSTR  = 392;  // cbar[b][h][i]
static constexpr int OSTR  = 392;  // attention out (bf16)
static constexpr int O2STR = 388;  // fc out (f32)
static constexpr int XSTR  = 520;  // [ln_out | src] (512 cols)
static constexpr int HSTR  = 136;  // hidden (128 cols)

// LDS byte offsets
static constexpr int KBUF_B = 0;        // 4*32*392*2  = 100352
static constexpr int KT_B   = 100352;   // 4*384*34*2  = 104448
static constexpr int QIN_B  = 204800;   // 16*392*2    =  12544
static constexpr int QH_B   = 217344;   // 16*392*2    =  12544
static constexpr int U_B    = 229888;   // 32*384*2    =  24576
static constexpr int SC_B   = 254464;   // 4*2*32*4    =   1024
static constexpr int AB_B   = 255488;   // 4*16*32*2   =   4096
static constexpr int CB_B   = 259584;   // 32*392*2    =  25088
static constexpr int LDS_BYTES = 284672;
// aliases inside the dead k-staging region after the chunk loop:
static constexpr int OUTB_B = 0;        // 16*392*2 = 12544
static constexpr int O2_B   = 12544;    // 16*388*4 = 24832
static constexpr int X_B    = 37376;    // 16*520*2 = 16640
static constexpr int HB_B   = 54016;    // 16*136*2 =  4352

// ---------------------------------------------------------------------------
// Generic K-loop:  C += A(16xK, LDS bf16) * B(Kx16), B rows indexed by n with
// contiguous (k,k+1) bf16 pairs.  A and B use the same contiguous k order:
// element s of lane half h holds k = 16h + s.
// ---------------------------------------------------------------------------
__device__ __forceinline__ v8f gemm_kloop(const bf16_t* A, int aStride, int aCol0,
                                          const bf16_t* B, int bStride, int nBase,
                                          int kTiles, int lane, v8f c) {
  const int r  = lane & 15;
  const int hi = lane >> 4;
  for (int kt = 0; kt < kTiles; ++kt) {
    Frag a, b;
    const int kb = kt * 32;
#pragma unroll
    for (int v = 0; v < 8; ++v) {
      const int ka = (v << 1) + (hi << 4);                        // A K-offset
      a.u[v] = *(const unsigned*)(A + r * aStride + aCol0 + kb + ka);
      const int kv = (v << 1) + (hi << 4);                        // B K-offset
      b.u[v] = *(const unsigned*)(B + (nBase + r) * bStride + kb + kv);
    }
    c = __builtin_amdgcn_wmma_f32_16x16x32_bf16(false, a.v, false, b.v,
                                                (short)0, c, false, false);
  }
  return c;
}

// ---------------------------------------------------------------------------
// Weight prep: f32 -> bf16 (w_k additionally transposed)
// ---------------------------------------------------------------------------
__global__ void __launch_bounds__(256) prep_weights(
    const float* __restrict__ wq, const float* __restrict__ wk,
    const float* __restrict__ wv, const float* __restrict__ fcw,
    const float* __restrict__ mg1, const float* __restrict__ mg2,
    bf16_t* __restrict__ ws) {
  int i = blockIdx.x * 256 + threadIdx.x;
  if (i >= WS_ELEMS) return;
  if (i < WKT_OFF) {
    ws[i] = (bf16_t)wq[i];
  } else if (i < WV_OFF) {
    int li = i - WKT_OFF; int ii = li / 384, j = li - ii * 384;
    ws[i] = (bf16_t)wk[j * 384 + ii];           // w_k^T[i][j] = w_k[j][i]
  } else if (i < FCW_OFF) {
    ws[i] = (bf16_t)wv[i - WV_OFF];
  } else if (i < MG1_OFF) {
    ws[i] = (bf16_t)fcw[i - FCW_OFF];
  } else if (i < MG2_OFF) {
    ws[i] = (bf16_t)mg1[i - MG1_OFF];
  } else {
    ws[i] = (bf16_t)mg2[i - MG2_OFF];
  }
}

// ---------------------------------------------------------------------------
// Fused kernel: 1 workgroup (256 thr = 8 waves) handles 16 batch elements.
// ---------------------------------------------------------------------------
__global__ void __launch_bounds__(256) tgat_fused(
    const float* __restrict__ src,  const float* __restrict__ src_t,
    const float* __restrict__ seq,  const float* __restrict__ seq_t,
    const float* __restrict__ seq_e,
    const unsigned char* __restrict__ mask,   // jax bool -> 1 byte/elem
    const bf16_t* __restrict__ ws,
    const float* __restrict__ fc_b, const float* __restrict__ ln_g,
    const float* __restrict__ ln_b, const float* __restrict__ mg_b1,
    const float* __restrict__ mg_b2,
    float* __restrict__ out) {
  extern __shared__ char smem[];
  bf16_t* kbuf  = (bf16_t*)(smem + KBUF_B);
  bf16_t* ktb   = (bf16_t*)(smem + KT_B);
  bf16_t* qin   = (bf16_t*)(smem + QIN_B);
  bf16_t* qh    = (bf16_t*)(smem + QH_B);
  bf16_t* ub    = (bf16_t*)(smem + U_B);
  float*  scbuf = (float*) (smem + SC_B);
  bf16_t* abuf  = (bf16_t*)(smem + AB_B);
  bf16_t* cbar  = (bf16_t*)(smem + CB_B);
  bf16_t* outb  = (bf16_t*)(smem + OUTB_B);
  float*  o2    = (float*) (smem + O2_B);
  bf16_t* xb    = (bf16_t*)(smem + X_B);
  bf16_t* hb    = (bf16_t*)(smem + HB_B);

  const int t     = threadIdx.x;
  const int lane  = t & 31;
  const int w     = t >> 5;
  const int r     = lane & 15;
  const int hi    = lane >> 4;
  const int group = blockIdx.x;        // 512 groups of 16 batches

  // ---- build Q input [src | 0 | src_t] (bf16) + zero attn A-matrix pad ----
  for (int it = 0; it < 24; ++it) {
    int id = t + 256 * it;             // 16*384
    int b = id / 384, j = id - b * 384;
    int bg = group * 16 + b;
    float v = (j < 128) ? src[bg * D_ + j]
                        : ((j < 256) ? 0.f : src_t[bg * D_ + (j - 256)]);
    qin[b * QSTR + j] = (bf16_t)v;
  }
  for (int it = 0; it < 8; ++it) {     // zero abuf (rows >=2 stay zero)
    int id = t + 256 * it;             // 4*16*32 = 2048
    abuf[id] = (bf16_t)0.f;
  }
  __syncthreads();

  // ---- G1: qh = qin @ w_q^T   (16x384) ----
  for (int idx = 0; idx < 3; ++idx) {
    int nt = w + 8 * idx;              // 0..23
    v8f cc = {};
    cc = gemm_kloop(qin, QSTR, 0, ws + WQ_OFF, 384, nt * 16, 12, lane, cc);
    int col = nt * 16 + r;
#pragma unroll
    for (int rr = 0; rr < 8; ++rr)
      qh[(rr + (hi << 3)) * QSTR + col] = (bf16_t)cc[rr];
  }
  __syncthreads();

  // ---- G2: u[b][h][i] = sum_j qh[b][h*192+j] * w_k[h*192+j][i] ----
  for (int idx = 0; idx < 6; ++idx) {
    int tid = w + 8 * idx;             // 0..47
    int h = tid / 24, nt = tid - h * 24;
    v8f cc = {};
    cc = gemm_kloop(qh, QSTR, h * DK_, ws + WKT_OFF + h * DK_, 384,
                    nt * 16, 6, lane, cc);
    int col = nt * 16 + r;
#pragma unroll
    for (int rr = 0; rr < 8; ++rr) {
      int b = rr + (hi << 3);
      ub[(b * 2 + h) * USTR + col] = (bf16_t)cc[rr];
    }
  }
  __syncthreads();

  // ---- stream k = [seq|seq_e|seq_t] in 4 chunks of 4 batches ----
  for (int c = 0; c < 4; ++c) {
    // load + convert to bf16 (row-major and transposed)
    for (int it = 0; it < 48; ++it) {
      int id = t + 256 * it;           // 4*32*96 float4's
      int q = id / 3072; int rem = id - q * 3072;
      int n = rem / 96;  int f = rem - n * 96;
      int seg = f >> 5;  int fo = (f & 31) << 2;
      int bg = group * 16 + c * 4 + q;
      const float* gp = (seg == 0 ? seq : (seg == 1 ? seq_e : seq_t)) +
                        ((size_t)(bg * N_ + n)) * D_ + fo;
      float4 val = *(const float4*)gp;
      int ib = seg * 128 + fo;
      bf16_t* kd = kbuf + (q * 32 + n) * KSTR + ib;
      kd[0] = (bf16_t)val.x; kd[1] = (bf16_t)val.y;
      kd[2] = (bf16_t)val.z; kd[3] = (bf16_t)val.w;
      bf16_t* kt0 = ktb + (q * 384 + ib) * KTSTR + n;
      kt0[0]         = (bf16_t)val.x; kt0[KTSTR]     = (bf16_t)val.y;
      kt0[2 * KTSTR] = (bf16_t)val.z; kt0[3 * KTSTR] = (bf16_t)val.w;
    }
    __syncthreads();

    // scores = k @ u  (WMMA; B has 2 real cols, lanes n>=2 produce junk)
    {
      int q = w >> 1, mt = w & 1, bl = c * 4 + q;
      const bf16_t* Ab = kbuf + (q * 32 + mt * 16) * KSTR;
      const bf16_t* Ub = ub + (bl * 2) * USTR;
      int hsel = (r < 2) ? r : 1;
      v8f cc = {};
      for (int kt = 0; kt < 12; ++kt) {
        Frag a, b; const int kb = kt * 32;
#pragma unroll
        for (int v = 0; v < 8; ++v) {
          int ka = (v << 1) + (hi << 4);
          a.u[v] = *(const unsigned*)(Ab + r * KSTR + kb + ka);
          int kv = (v << 1) + (hi << 4);
          b.u[v] = *(const unsigned*)(Ub + hsel * USTR + kb + kv);
        }
        cc = __builtin_amdgcn_wmma_f32_16x16x32_bf16(false, a.v, false, b.v,
                                                     (short)0, cc, false, false);
      }
      if (r < 2) {
        const float invs = 0.07216878364870322f;  // 1/sqrt(192)
#pragma unroll
        for (int rr = 0; rr < 8; ++rr) {
          int nidx = mt * 16 + rr + (hi << 3);
          scbuf[(q * 2 + r) * 32 + nidx] = cc[rr] * invs;
        }
      }
    }
    __syncthreads();

    // masked softmax (8 rows, serial per thread) + attn_w output
    if (t < 8) {
      int q = t >> 1, h = t & 1, bl = c * 4 + q, bg = group * 16 + bl;
      float* srow = scbuf + (q * 2 + h) * 32;
      float mx = -3.4e38f;
      for (int n = 0; n < 32; ++n) {
        float s = srow[n];
        if (mask[bg * N_ + n]) s = -1e10f;
        srow[n] = s; mx = fmaxf(mx, s);
      }
      float sum = 0.f;
      for (int n = 0; n < 32; ++n) { float e = __expf(srow[n] - mx); srow[n] = e; sum += e; }
      float inv = 1.f / sum;
      float* aw = out + (size_t)B_ * D_ + ((size_t)h * B_ + bg) * N_;
      for (int n = 0; n < 32; ++n) {
        float a = srow[n] * inv;
        abuf[(q * 16 + h) * 32 + n] = (bf16_t)a;
        aw[n] = a;
      }
    }
    __syncthreads();

    // cbar[b][h][i] = attn[h] @ k   (WMMA; A rows>=2 are zero)
    for (int idx = 0; idx < 12; ++idx) {
      int jb = w + 8 * idx;            // 0..95
      int q = jb / 24, nt = jb - q * 24, bl = c * 4 + q;
      Frag a, b;
#pragma unroll
      for (int v = 0; v < 8; ++v) {
        int ka = (v << 1) + (hi << 4);
        a.u[v] = *(const unsigned*)(abuf + (q * 16 + r) * 32 + ka);
        int kv = (v << 1) + (hi << 4);
        b.u[v] = *(const unsigned*)(ktb + (q * 384 + nt * 16 + r) * KTSTR + kv);
      }
      v8f cc = {};
      cc = __builtin_amdgcn_wmma_f32_16x16x32_bf16(false, a.v, false, b.v,
                                                   (short)0, cc, false, false);
      if (hi == 0) {
        int col = nt * 16 + r;
        cbar[(bl * 2 + 0) * CSTR + col] = (bf16_t)cc[0];   // row 0 = head 0
        cbar[(bl * 2 + 1) * CSTR + col] = (bf16_t)cc[1];   // row 1 = head 1
      }
    }
    __syncthreads();                   // before next chunk reuses kbuf/ktb
  }

  // ---- G3: out[b][h*192+j] = cbar[b][h] @ w_v_head^T ----
  for (int idx = 0; idx < 3; ++idx) {
    int tid = w + 8 * idx;             // 0..23
    int h = tid / 12, nt = tid - h * 12;
    v8f cc = {};
    cc = gemm_kloop(cbar + h * CSTR, 2 * CSTR, 0,
                    ws + WV_OFF + (h * DK_) * 384, 384, nt * 16, 12, lane, cc);
    int col = h * DK_ + nt * 16 + r;
#pragma unroll
    for (int rr = 0; rr < 8; ++rr)
      outb[(rr + (hi << 3)) * OSTR + col] = (bf16_t)cc[rr];
  }
  __syncthreads();

  // ---- G4: o2 = outb @ fc_w^T + fc_b ----
  for (int idx = 0; idx < 3; ++idx) {
    int nt = w + 8 * idx;
    v8f cc = {};
    cc = gemm_kloop(outb, OSTR, 0, ws + FCW_OFF, 384, nt * 16, 12, lane, cc);
    int col = nt * 16 + r;
    float bias = fc_b[col];
#pragma unroll
    for (int rr = 0; rr < 8; ++rr)
      o2[(rr + (hi << 3)) * O2STR + col] = cc[rr] + bias;
  }
  __syncthreads();

  // ---- LayerNorm(384) -> X[:,0:384]; X[:,384:512] = src (bf16) ----
  {
    int row = t >> 4, cp = t & 15;
    float s = 0.f, s2 = 0.f;
    for (int j = cp; j < 384; j += 16) { float v = o2[row * O2STR + j]; s += v; s2 += v * v; }
    for (int m2 = 8; m2 >= 1; m2 >>= 1) { s += __shfl_xor(s, m2, 16); s2 += __shfl_xor(s2, m2, 16); }
    float mu = s * (1.f / 384.f);
    float var = s2 * (1.f / 384.f) - mu * mu;
    float rstd = rsqrtf(var + 1e-5f);
    for (int j = cp; j < 384; j += 16) {
      float v = (o2[row * O2STR + j] - mu) * rstd * ln_g[j] + ln_b[j];
      xb[row * XSTR + j] = (bf16_t)v;
    }
  }
  for (int it = 0; it < 8; ++it) {
    int id = t + 256 * it;             // 16*128
    int b = id >> 7, i = id & 127;
    xb[b * XSTR + 384 + i] = qin[b * QSTR + i];   // src already bf16 in qin
  }
  __syncthreads();

  // ---- G5: h = relu(X @ mg_w1^T + mg_b1) ----
  {
    int nt = w;                        // 8 N-tiles, 8 waves
    v8f cc = {};
    cc = gemm_kloop(xb, XSTR, 0, ws + MG1_OFF, 512, nt * 16, 16, lane, cc);
    int col = nt * 16 + r;
    float bias = mg_b1[col];
#pragma unroll
    for (int rr = 0; rr < 8; ++rr)
      hb[(rr + (hi << 3)) * HSTR + col] = (bf16_t)fmaxf(cc[rr] + bias, 0.f);
  }
  __syncthreads();

  // ---- G6: merged = h @ mg_w2^T + mg_b2 -> global ----
  {
    int nt = w;
    v8f cc = {};
    cc = gemm_kloop(hb, HSTR, 0, ws + MG2_OFF, 128, nt * 16, 4, lane, cc);
    int col = nt * 16 + r;
    float bias = mg_b2[col];
#pragma unroll
    for (int rr = 0; rr < 8; ++rr) {
      int bg = group * 16 + rr + (hi << 3);
      out[(size_t)bg * D_ + col] = cc[rr] + bias;
    }
  }
}

// ---------------------------------------------------------------------------
extern "C" void kernel_launch(void* const* d_in, const int* in_sizes, int n_in,
                              void* d_out, int out_size, void* d_ws, size_t ws_size,
                              hipStream_t stream) {
  const float* src    = (const float*)d_in[0];
  const float* src_t  = (const float*)d_in[1];
  const float* seq    = (const float*)d_in[2];
  const float* seq_t  = (const float*)d_in[3];
  const float* seq_e  = (const float*)d_in[4];
  // d_in[5] = time_diff (unused by the reference model)
  const unsigned char* mask = (const unsigned char*)d_in[6];
  const float* w_q    = (const float*)d_in[7];
  const float* w_k    = (const float*)d_in[8];
  const float* w_v    = (const float*)d_in[9];
  const float* fc_w   = (const float*)d_in[10];
  const float* fc_b   = (const float*)d_in[11];
  const float* ln_g   = (const float*)d_in[12];
  const float* ln_b   = (const float*)d_in[13];
  const float* mg_w1  = (const float*)d_in[14];
  const float* mg_b1  = (const float*)d_in[15];
  const float* mg_w2  = (const float*)d_in[16];
  const float* mg_b2  = (const float*)d_in[17];
  float* out = (float*)d_out;
  bf16_t* ws = (bf16_t*)d_ws;

  (void)in_sizes; (void)n_in; (void)out_size; (void)ws_size;

  hipFuncSetAttribute((const void*)tgat_fused,
                      hipFuncAttributeMaxDynamicSharedMemorySize, LDS_BYTES);

  prep_weights<<<(WS_ELEMS + 255) / 256, 256, 0, stream>>>(
      w_q, w_k, w_v, fc_w, mg_w1, mg_w2, ws);

  tgat_fused<<<B_ / 16, 256, LDS_BYTES, stream>>>(
      src, src_t, seq, seq_t, seq_e, mask, ws,
      fc_b, ln_g, ln_b, mg_b1, mg_b2, out);
}
